// Attention_46119358824792
// MI455X (gfx1250) — compile-verified, matrix operand order checked by reading, not run on hardware
//
#include <hip/hip_runtime.h>
#include <hip/hip_bf16.h>

typedef __attribute__((ext_vector_type(16))) _Float16 v16h;
typedef __attribute__((ext_vector_type(8)))  _Float16 v8h;
typedef __attribute__((ext_vector_type(8)))  float    v8f;
typedef __attribute__((ext_vector_type(4)))  unsigned int u32x4;
typedef __attribute__((ext_vector_type(8)))  int          i32x8;
typedef __attribute__((ext_vector_type(4)))  int          i32x4;

#define BB 4
#define NN 2048
#define CC 768
#define HH 12
#define DD 64
#define SCALE 0.125f
#define MASK_NEG -100000.0f

// Tensor Data Mover availability (device pass only; arity differs per toolchain)
#if defined(__HIP_DEVICE_COMPILE__) && \
    __has_builtin(__builtin_amdgcn_tensor_load_to_lds) && \
    __has_builtin(__builtin_amdgcn_s_wait_tensorcnt)
#define HAVE_TDM 1
#else
#define HAVE_TDM 0
#endif

// ---------------------------------------------------------------------------
// WMMA helpers
// ---------------------------------------------------------------------------
static __device__ inline v8f wmma16(v16h a, v16h b, v8f c) {
  return __builtin_amdgcn_wmma_f32_16x16x32_f16(
      /*neg_a=*/false, a, /*neg_b=*/false, b,
      /*c_mod=*/(short)0, c, /*reuse_a=*/false, /*reuse_b=*/false);
}

// Load one 16-half fragment slice: two contiguous 8-half (16B) chunks at
// p[0..7] and p[16..23].  Caller folds the per-lane kbase into p.
static __device__ inline v16h ldfrag(const _Float16* p) {
  v8h lo = *(const v8h*)(p);
  v8h hi = *(const v8h*)(p + 16);
  v16h r;
#pragma unroll
  for (int i = 0; i < 8; ++i) { r[i] = lo[i]; r[i + 8] = hi[i]; }
  return r;
}

#if HAVE_TDM
// Issue a TDM copy of a [rows=32 x cols=64] f16 row-major tile from global
// memory into LDS.  D# packing per CDNA5 ISA ch.8: group0 holds count /
// lds_addr / global_addr / type=2; group1 holds data_size, dims, strides.
static __device__ inline void tdm_load_tile_2d(const _Float16* gptr,
                                               unsigned lds_addr,
                                               unsigned rows_remaining) {
  unsigned long long ga = (unsigned long long)(size_t)gptr;
  u32x4 g0 = {
      1u,                                                   // count=1 (valid)
      lds_addr,                                             // lds_addr bytes
      (unsigned)ga,                                         // global_addr lo
      (unsigned)((ga >> 32) & 0x01FFFFFFu) | (2u << 30)     // addr hi | type=2
  };
  i32x8 g1 = {
      (int)(1u << 16),                   // [17:16] data_size = 1 (2 bytes)
      (int)(64u << 16),                  // [79:48] tensor_dim0 = 64 (lo16)
      (int)(rows_remaining << 16),       // [111:80] tensor_dim1 (lo16)
      (int)(64u << 16),                  // [127:112] tile_dim0 = 64
      32,                                // [143:128] tile_dim1 = 32 rows
      64,                                // [207:160] tensor_dim0_stride = 64
      0, 0                               // dim1_stride unused (2-D tile)
  };
  i32x4 z4 = {0, 0, 0, 0};
#if __clang_major__ >= 23
  i32x8 z8 = {0, 0, 0, 0, 0, 0, 0, 0};
  __builtin_amdgcn_tensor_load_to_lds(g0, g1, z4, z4, z8, 0);
#else
  __builtin_amdgcn_tensor_load_to_lds(g0, g1, z4, z4, 0);
#endif
}
#endif

// ---------------------------------------------------------------------------
// fp32 -> f16 conversion (grid-stride)
// ---------------------------------------------------------------------------
__global__ void cvt_f32_f16(const float* __restrict__ src,
                            _Float16* __restrict__ dst, int n) {
  for (int i = blockIdx.x * blockDim.x + threadIdx.x; i < n;
       i += gridDim.x * blockDim.x)
    dst[i] = (_Float16)src[i];
}

// ---------------------------------------------------------------------------
// Generic Y = X[M,K] * W[Nout,K]^T  GEMM, wave computes a 32x32 tile.
// MODE 0: scatter into Q heads  [B,H,N,D] f16          (out16)
// MODE 1: scatter into K/V heads[B,H,N,D] f16          (out16 = K, out16b = V)
// MODE 2: plain fp32 row-major + bias                  (outf)
// ---------------------------------------------------------------------------
template <int MODE>
__global__ __launch_bounds__(256) void gemm_kernel(
    const _Float16* __restrict__ X, const _Float16* __restrict__ W,
    _Float16* __restrict__ out16, _Float16* __restrict__ out16b,
    float* __restrict__ outf, const float* __restrict__ bias,
    int M, int Nout, int Kdim) {
  const int lane = threadIdx.x & 31;
  const int wave = threadIdx.x >> 5;
  const int wm = wave >> 2;                 // 0..1
  const int wn = wave & 3;                  // 0..3
  const int m0 = blockIdx.x * 64 + wm * 32;
  const int n0 = blockIdx.y * 128 + wn * 32;
  const int l15 = lane & 15;
  const int kb  = (lane & 16) ? 8 : 0;

  const _Float16* xr0 = X + (size_t)(m0 + l15) * Kdim;
  const _Float16* xr1 = X + (size_t)(m0 + 16 + l15) * Kdim;
  const _Float16* wr0 = W + (size_t)(n0 + l15) * Kdim;
  const _Float16* wr1 = W + (size_t)(n0 + 16 + l15) * Kdim;

  v8f acc00 = {}, acc01 = {}, acc10 = {}, acc11 = {};
  for (int k = 0; k < Kdim; k += 32) {
    v16h a0 = ldfrag(xr0 + k + kb);
    v16h a1 = ldfrag(xr1 + k + kb);
    v16h b0 = ldfrag(wr0 + k + kb);
    v16h b1 = ldfrag(wr1 + k + kb);
    acc00 = wmma16(a0, b0, acc00);
    acc01 = wmma16(a0, b1, acc01);
    acc10 = wmma16(a1, b0, acc10);
    acc11 = wmma16(a1, b1, acc11);
  }

  const int g = lane >> 4;
#pragma unroll
  for (int r = 0; r < 8; ++r) {
    int mm[2] = {m0 + r + 8 * g, m0 + 16 + r + 8 * g};
    int nn2[2] = {n0 + l15, n0 + 16 + l15};
    float vv[2][2] = {{acc00[r], acc01[r]}, {acc10[r], acc11[r]}};
#pragma unroll
    for (int i = 0; i < 2; ++i) {
#pragma unroll
      for (int j = 0; j < 2; ++j) {
        int m = mm[i], n = nn2[j];
        float val = vv[i][j];
        if (MODE == 0) {  // Q heads
          int b = m >> 11, nr = m & (NN - 1);
          int h = n >> 6, d = n & 63;
          out16[(((size_t)b * HH + h) * NN + nr) * DD + d] = (_Float16)val;
        } else if (MODE == 1) {  // K/V heads: n in [0,2C)
          int b = m >> 11, nr = m & (NN - 1);
          int s = n / CC, c = n % CC;
          int h = c >> 6, d = c & 63;
          _Float16* dst = s ? out16b : out16;
          dst[(((size_t)b * HH + h) * NN + nr) * DD + d] = (_Float16)val;
        } else {  // fp32 + bias
          outf[(size_t)m * Nout + n] = val + bias[n];
        }
      }
    }
  }
}

// ---------------------------------------------------------------------------
// Flash attention: grid (N/128, B*H), 256 threads = 8 waves, each wave owns
// 16 query rows.  The shared K tile (32 keys x 64 d, row-major) is DMA'd
// into LDS by the Tensor Data Mover (wave 0, TENSORcnt); the V tile is
// cooperatively transposed into LDS so all fragment reads are aligned 16B
// ds_load_b128s.  Online softmax in fp32 on the S registers.
// ---------------------------------------------------------------------------
__global__ __launch_bounds__(256, 1) void attn_kernel(
    const _Float16* __restrict__ Q, const _Float16* __restrict__ K,
    const _Float16* __restrict__ V, const float* __restrict__ mask,
    _Float16* __restrict__ ao) {
  __shared__ __align__(16) _Float16 klds[32 * 64];     // [key][d]   4 KB
  __shared__ __align__(16) _Float16 vt[64 * 32];       // [d][key]   4 KB
  __shared__ __align__(16) _Float16 plds[8][16 * 32];  // P slabs    8 KB

  const int tid = threadIdx.x;
  const int lane = tid & 31;
  const int wave = tid >> 5;
  const int bh = blockIdx.y;            // 0..47
  const int b = bh / HH, h = bh % HH;
  const int q0 = blockIdx.x * 128 + wave * 16;
  const int l15 = lane & 15;
  const int g = lane >> 4;
  const int kb = g ? 8 : 0;

  const _Float16* Qbh = Q + (size_t)bh * NN * DD;
  const _Float16* Kbh = K + (size_t)bh * NN * DD;
  const _Float16* Vbh = V + (size_t)bh * NN * DD;
  const float* maskb = mask + (size_t)b * NN * NN;

  const _Float16* qrow = Qbh + (size_t)(q0 + l15) * DD;
  const v16h qa0 = ldfrag(qrow + kb);        // d 0..31
  const v16h qa1 = ldfrag(qrow + 32 + kb);   // d 32..63

  // cooperative staging coordinates: thread -> (key row, 8-wide d chunk)
  const int skey = tid >> 3;           // 0..31
  const int sdc  = (tid & 7) * 8;      // 0,8,...,56

  float Mrun[8], Lrun[8];
#pragma unroll
  for (int r = 0; r < 8; ++r) { Mrun[r] = -1e30f; Lrun[r] = 0.0f; }
  v8f o0 = {}, o1 = {}, o2 = {}, o3 = {};

  _Float16* slab = &plds[wave][0];

  for (int kt = 0; kt < NN; kt += 32) {
    // ---- phase 1: fetch the shared K/V tile ----
    const _Float16* vgp = Vbh + (size_t)(kt + skey) * DD + sdc;
    v8h vreg = *(const v8h*)vgp;
#if !HAVE_TDM
    const _Float16* kgp = Kbh + (size_t)(kt + skey) * DD + sdc;
    v8h kreg = *(const v8h*)kgp;
#endif
    if (kt + 32 < NN) {  // prefetch next V tile (global_prefetch_b8)
      __builtin_prefetch(vgp + 32 * DD, 0, 1);
    }

    __syncthreads();  // previous iteration's LDS consumers are done
#if HAVE_TDM
    if (wave == 0) {   // TDM: DMA the K tile straight into LDS
      tdm_load_tile_2d(Kbh + (size_t)kt * DD,
                       (unsigned)(size_t)(void*)klds,
                       (unsigned)(NN - kt));
    }
#else
    *(v8h*)(klds + skey * DD + sdc) = kreg;  // [key][d] row-major
#endif
#pragma unroll
    for (int j = 0; j < 8; ++j)              // transpose V -> vt[d][key]
      vt[(sdc + j) * 32 + skey] = vreg[j];
#if HAVE_TDM
    if (wave == 0) __builtin_amdgcn_s_wait_tensorcnt(0);
#endif
    __syncthreads();

    // ---- phase 2: S = Q K^T for 32 keys (LDS-resident K tile) ----
    const _Float16* kr0 = klds + l15 * DD;          // key = l15
    const _Float16* kr1 = klds + (16 + l15) * DD;   // key = 16 + l15
    v8f s0 = {}, s1 = {};
    s0 = wmma16(qa0, ldfrag(kr0 + kb), s0);
    s0 = wmma16(qa1, ldfrag(kr0 + 32 + kb), s0);
    s1 = wmma16(qa0, ldfrag(kr1 + kb), s1);
    s1 = wmma16(qa1, ldfrag(kr1 + 32 + kb), s1);

    // ---- phase 3: scale + mask + online softmax, in place on s0/s1 ----
#pragma unroll
    for (int r = 0; r < 8; ++r) {
      int qr = q0 + r + 8 * g;
      const float* mrow = maskb + (size_t)qr * NN + kt;
      s0[r] = s0[r] * SCALE + mrow[l15] * MASK_NEG;
      s1[r] = s1[r] * SCALE + mrow[16 + l15] * MASK_NEG;
    }
#pragma unroll
    for (int r = 0; r < 8; ++r) {
      float mx = fmaxf(s0[r], s1[r]);
#pragma unroll
      for (int xm = 1; xm <= 8; xm <<= 1)
        mx = fmaxf(mx, __shfl_xor(mx, xm, 32));
      float mnew = fmaxf(Mrun[r], mx);
      float so = __expf(Mrun[r] - mnew);
      s0[r] = __expf(s0[r] - mnew);
      s1[r] = __expf(s1[r] - mnew);
      float rs = s0[r] + s1[r];
#pragma unroll
      for (int xm = 1; xm <= 8; xm <<= 1)
        rs += __shfl_xor(rs, xm, 32);
      Lrun[r] = Lrun[r] * so + rs;
      Mrun[r] = mnew;
      o0[r] *= so; o1[r] *= so; o2[r] *= so; o3[r] *= so;
    }

    // ---- phase 4: P (C-frag layout) -> A-frag layout via LDS slab ----
#pragma unroll
    for (int r = 0; r < 8; ++r) {
      int row = r + 8 * g;
      slab[row * 32 + l15]      = (_Float16)s0[r];
      slab[row * 32 + 16 + l15] = (_Float16)s1[r];
    }
    __syncthreads();
    v16h pa = ldfrag(slab + l15 * 32 + kb);

    // ---- phase 5: O += P V, B-frags from transposed LDS V tile ----
    o0 = wmma16(pa, ldfrag(vt + (0 * 16 + l15) * 32 + kb), o0);
    o1 = wmma16(pa, ldfrag(vt + (1 * 16 + l15) * 32 + kb), o1);
    o2 = wmma16(pa, ldfrag(vt + (2 * 16 + l15) * 32 + kb), o2);
    o3 = wmma16(pa, ldfrag(vt + (3 * 16 + l15) * 32 + kb), o3);
  }

  // ---- epilogue: normalize and scatter into [B,N,C] f16 ----
#pragma unroll
  for (int r = 0; r < 8; ++r) {
    float inv = 1.0f / Lrun[r];
    int qr = q0 + r + 8 * g;
    _Float16* dst = ao + ((size_t)b * NN + qr) * CC + h * DD;
    dst[0 * 16 + l15] = (_Float16)(o0[r] * inv);
    dst[1 * 16 + l15] = (_Float16)(o1[r] * inv);
    dst[2 * 16 + l15] = (_Float16)(o2[r] * inv);
    dst[3 * 16 + l15] = (_Float16)(o3[r] * inv);
  }
}

// ---------------------------------------------------------------------------
// Host launcher
// ---------------------------------------------------------------------------
extern "C" void kernel_launch(void* const* d_in, const int* in_sizes, int n_in,
                              void* d_out, int out_size, void* d_ws,
                              size_t ws_size, hipStream_t stream) {
  const float* x    = (const float*)d_in[0];   // [B,N,C]
  const float* mask = (const float*)d_in[1];   // [B,N,N]
  const float* Wq   = (const float*)d_in[2];   // [C,C]
  const float* Wkv  = (const float*)d_in[3];   // [2C,C]
  const float* Wp   = (const float*)d_in[4];   // [C,C]
  const float* bp   = (const float*)d_in[5];   // [C]
  float* out = (float*)d_out;                  // [B,N,C]

  // workspace carve-out (256B aligned)
  char* ws = (char*)d_ws;
  size_t off = 0;
  auto carve = [&](size_t bytes) {
    char* p = ws + off;
    off += (bytes + 255) & ~(size_t)255;
    return p;
  };
  const size_t nX  = (size_t)BB * NN * CC;       // 6.29M
  const size_t nWq = (size_t)CC * CC;
  const size_t nWkv = (size_t)2 * CC * CC;
  const size_t nHead = (size_t)BB * HH * NN * DD;

  _Float16* xh   = (_Float16*)carve(nX * 2);
  _Float16* wqh  = (_Float16*)carve(nWq * 2);
  _Float16* wkvh = (_Float16*)carve(nWkv * 2);
  _Float16* wph  = (_Float16*)carve(nWq * 2);
  _Float16* qh   = (_Float16*)carve(nHead * 2);
  _Float16* kh   = (_Float16*)carve(nHead * 2);
  _Float16* vh   = (_Float16*)carve(nHead * 2);
  _Float16* ao   = (_Float16*)carve(nX * 2);
  (void)ws_size;

  // 1) fp32 -> f16 conversions
  cvt_f32_f16<<<2048, 256, 0, stream>>>(x, xh, (int)nX);
  cvt_f32_f16<<<1024, 256, 0, stream>>>(Wq, wqh, (int)nWq);
  cvt_f32_f16<<<1024, 256, 0, stream>>>(Wkv, wkvh, (int)nWkv);
  cvt_f32_f16<<<1024, 256, 0, stream>>>(Wp, wph, (int)nWq);

  const int M = BB * NN;  // 8192

  // 2) Q projection -> [B,H,N,D]
  gemm_kernel<0><<<dim3(M / 64, CC / 128), 256, 0, stream>>>(
      xh, wqh, qh, nullptr, nullptr, nullptr, M, CC, CC);

  // 3) KV projection -> K,V [B,H,N,D]
  gemm_kernel<1><<<dim3(M / 64, (2 * CC) / 128), 256, 0, stream>>>(
      xh, wkvh, kh, vh, nullptr, nullptr, M, 2 * CC, CC);

  // 4) flash attention -> ao [B,N,C] f16
  attn_kernel<<<dim3(NN / 128, BB * HH), 256, 0, stream>>>(qh, kh, vh, mask, ao);

  // 5) output projection + bias -> fp32 d_out
  gemm_kernel<2><<<dim3(M / 64, CC / 128), 256, 0, stream>>>(
      ao, wph, nullptr, nullptr, out, bp, M, CC, CC);
}
